// SmallWorldVideoAttention_84078279786897
// MI455X (gfx1250) — matrix operand order repaired and checked by reading, not compiled, not run on hardware
//
#include <hip/hip_runtime.h>
#include <hip/hip_bf16.h>

// ---------------------------------------------------------------------------
// SmallWorldVideoAttention for MI455X (gfx1250, wave32, WMMA).
// B=1, L=8192, C=1024, H=16, D=64, T=8, S=1024, 16 shifts.
// GEMMs (4x 8192x1024x1024) via v_wmma_f32_16x16x32_bf16; epilogues in f32.
// Weights packed TRANSPOSED once; GEMM staging uses CDNA5 async global->LDS
// (GLOBAL_LOAD_ASYNC_TO_LDS_B128 + s_wait_asynccnt) — no VGPR round-trip.
// ---------------------------------------------------------------------------

typedef __bf16 bf16;
typedef __attribute__((ext_vector_type(16))) __bf16 v16bf;
typedef __attribute__((ext_vector_type(8)))  __bf16 v8bf;
typedef __attribute__((ext_vector_type(8)))  float  v8f;

#define LL 8192      // tokens
#define CC 1024      // channels
#define NH 16        // heads
#define HD 64        // head dim
#define TT 8         // frames
#define SS 1024      // spatial per frame
#define NQKV 3072    // 3*CC
#define NSH 16       // shifts

#define USE_ASYNC_LDS 1   // flip to 0 to fall back to load->ds_store staging

#if USE_ASYNC_LDS
// Async DMA one 16B chunk from global memory directly into LDS (ASYNCcnt).
// lds_off: wave-relative LDS byte address (low 32 bits of a __shared__ ptr).
__device__ __forceinline__ void async_ld_b128(unsigned lds_off, const bf16* g) {
  unsigned long long ga = (unsigned long long)(size_t)g;
  asm volatile("global_load_async_to_lds_b128 %0, %1, off"
               :: "v"(lds_off), "v"(ga) : "memory");
}
__device__ __forceinline__ void wait_async0() {
  asm volatile("s_wait_asynccnt 0x0" ::: "memory");
}
#endif

// ---------------- conversion / packing kernels ----------------

__global__ __launch_bounds__(256) void conv_f32_to_bf16(const float* __restrict__ in,
                                                        bf16* __restrict__ out, int n) {
  int i0 = (blockIdx.x * 256 + threadIdx.x) * 4;
  if (i0 + 3 < n) {
#pragma unroll
    for (int j = 0; j < 4; ++j) out[i0 + j] = (bf16)in[i0 + j];
  }
}

// Pack Wq|Wk|Wv ([K=1024][N=1024] each, row-major) into transposed bf16
// WqkvT [N=3072][K=1024], K-contiguous.
__global__ __launch_bounds__(256) void pack_wqkvT(const float* __restrict__ Wq,
                                                  const float* __restrict__ Wk,
                                                  const float* __restrict__ Wv,
                                                  bf16* __restrict__ WT) {
  int idx = blockIdx.x * 256 + threadIdx.x;        // 0 .. 3*1024*1024-1
  int which = idx >> 20;                           // 0,1,2
  int rem = idx & ((1 << 20) - 1);
  int k = rem >> 10, n = rem & 1023;
  const float* src = (which == 0) ? Wq : (which == 1) ? Wk : Wv;
  WT[(size_t)(which * CC + n) * CC + k] = (bf16)src[rem];
}

// Pack Wo ([K=1024][N=1024] row-major) into transposed bf16 WoT [N][K].
__global__ __launch_bounds__(256) void pack_woT(const float* __restrict__ Wo,
                                                bf16* __restrict__ WT) {
  int idx = blockIdx.x * 256 + threadIdx.x;        // 0 .. 1024*1024-1
  int k = idx >> 10, n = idx & 1023;
  WT[(size_t)n * CC + k] = (bf16)Wo[idx];
}

__global__ __launch_bounds__(256) void pack_bqkv(const float* __restrict__ bq,
                                                 const float* __restrict__ bk,
                                                 const float* __restrict__ bv,
                                                 float* __restrict__ bqkv) {
  int i = blockIdx.x * 256 + threadIdx.x;          // 0..3071
  int which = i >> 10;
  const float* src = (which == 0) ? bq : (which == 1) ? bk : bv;
  bqkv[i] = src[i & 1023];
}

// ---------------- WMMA bf16 GEMM: C[M,N] = A[M,K] @ BT[N,K]^T + bias -------
// Workgroup tile 256(M) x 64(N), 256 threads = 8 waves.
// Wave w owns rows [32w, 32w+32): 2 A fragments x 4 N-chunks = 8 WMMA / k-step.
// K stepped by 32; staging via async global->LDS b128 DMA.

#define KB 32
#define LDA_P 40   // LDS row stride (bf16 elems), 16B aligned (80 bytes)
#define LDB_P 40

__global__ __launch_bounds__(256) void gemm_bf16_wmma(const bf16* __restrict__ Ag,
                                                      const bf16* __restrict__ BTg,
                                                      const float* __restrict__ bias,
                                                      float* __restrict__ C,
                                                      int M, int N, int K) {
  __shared__ __align__(16) bf16 ldsA[256 * LDA_P];   // 20 KB
  __shared__ __align__(16) bf16 ldsB[64 * LDB_P];    //  5 KB

  const int tid  = threadIdx.x;
  const int lane = tid & 31;
  const int wave = tid >> 5;
  const int ntile = blockIdx.x * 64;
  const int mtile = blockIdx.y * 256;

  v8f acc[2][4];
#pragma unroll
  for (int p = 0; p < 2; ++p)
#pragma unroll
    for (int c = 0; c < 4; ++c)
#pragma unroll
      for (int r = 0; r < 8; ++r) acc[p][c][r] = 0.0f;

  const int lrow = lane & 15;
  const int kbA = (lane < 16) ? 0 : 8;    // A frag: K {kbA..kbA+7, kbA+16..kbA+23}
  const int kbB = (lane < 16) ? 0 : 8;    // B frag: K {kbB..kbB+7, kbB+16..kbB+23}

  const bf16* srcA = Ag + (size_t)(mtile + tid) * K;           // one row / thread
  const bf16* srcB = BTg + (size_t)(ntile + (tid >> 2)) * K + (tid & 3) * 8;
#if USE_ASYNC_LDS
  const unsigned ldsAoff = (unsigned)(size_t)&ldsA[tid * LDA_P];
  const unsigned ldsBoff = (unsigned)(size_t)&ldsB[(tid >> 2) * LDB_P + (tid & 3) * 8];
#endif

  for (int k0 = 0; k0 < K; k0 += KB) {
    __syncthreads();   // previous iteration's LDS reads complete
    {
      const bf16* s = srcA + k0;
#if USE_ASYNC_LDS
      // --- A tile: 256 rows x 32 cols; thread stages one row, 4x async b128 ---
      async_ld_b128(ldsAoff,      s);
      async_ld_b128(ldsAoff + 16, s + 8);
      async_ld_b128(ldsAoff + 32, s + 16);
      async_ld_b128(ldsAoff + 48, s + 24);
      // --- B tile: 64 rows (n) x 32 cols (k); one async b128 per thread ---
      async_ld_b128(ldsBoff, srcB + k0);
#else
      v8bf d0 = *(const v8bf*)(s);
      v8bf d1 = *(const v8bf*)(s + 8);
      v8bf d2 = *(const v8bf*)(s + 16);
      v8bf d3 = *(const v8bf*)(s + 24);
      bf16* l = &ldsA[tid * LDA_P];
      *(v8bf*)(l)      = d0;
      *(v8bf*)(l + 8)  = d1;
      *(v8bf*)(l + 16) = d2;
      *(v8bf*)(l + 24) = d3;
      v8bf db = *(const v8bf*)(srcB + k0);
      *(v8bf*)&ldsB[(tid >> 2) * LDB_P + (tid & 3) * 8] = db;
#endif
      if (k0 + KB < K)  // prefetch next A tile -> global_prefetch_b8
        __builtin_prefetch(s + KB, 0, 1);
    }
#if USE_ASYNC_LDS
    wait_async0();     // this thread's DMA into LDS is visible
#endif
    __syncthreads();   // whole tile staged

    // --- A fragments for this wave's two 16-row strips ---
    v16bf afrag[2];
#pragma unroll
    for (int p = 0; p < 2; ++p) {
      const bf16* ap = &ldsA[(32 * wave + 16 * p + lrow) * LDA_P + kbA];
      v8bf alo = *(const v8bf*)(ap);
      v8bf ahi = *(const v8bf*)(ap + 16);
      afrag[p] = __builtin_shufflevector(alo, ahi, 0, 1, 2, 3, 4, 5, 6, 7,
                                         8, 9, 10, 11, 12, 13, 14, 15);
    }
#pragma unroll
    for (int c = 0; c < 4; ++c) {
      const bf16* bp = &ldsB[(16 * c + lrow) * LDB_P + kbB];
      v8bf blo = *(const v8bf*)(bp);
      v8bf bhi = *(const v8bf*)(bp + 16);
      v16bf bfrag = __builtin_shufflevector(blo, bhi, 0, 1, 2, 3, 4, 5, 6, 7,
                                            8, 9, 10, 11, 12, 13, 14, 15);
      acc[0][c] = __builtin_amdgcn_wmma_f32_16x16x32_bf16(
          false, afrag[0], false, bfrag, (short)0, acc[0][c], false, false);
      acc[1][c] = __builtin_amdgcn_wmma_f32_16x16x32_bf16(
          false, afrag[1], false, bfrag, (short)0, acc[1][c], false, false);
    }
  }

  // --- epilogue: C[m][n] = acc + bias[n] ---
#pragma unroll
  for (int c = 0; c < 4; ++c) {
    int n = ntile + 16 * c + lrow;
    float b = bias[n];
#pragma unroll
    for (int p = 0; p < 2; ++p) {
#pragma unroll
      for (int r = 0; r < 8; ++r) {
        int m = mtile + 32 * wave + 16 * p + ((lane < 16) ? r : r + 8);
        C[(size_t)m * N + n] = acc[p][c][r] + b;
      }
    }
  }
}

// ---------------- RMS norm over inner=1024, in place on Q and K ------------

__global__ __launch_bounds__(256) void rms_kernel(float* __restrict__ qkv,
                                                  const float* __restrict__ qn_w,
                                                  const float* __restrict__ kn_w) {
  __shared__ float red[8];
  const int l = blockIdx.x;
  const int tid = threadIdx.x;
  const int lane = tid & 31, wave = tid >> 5;
#pragma unroll
  for (int which = 0; which < 2; ++which) {
    float* row = qkv + (size_t)l * NQKV + which * CC;
    const float* w = which ? kn_w : qn_w;
    int i0 = tid * 4;
    float v0 = row[i0], v1 = row[i0 + 1], v2 = row[i0 + 2], v3 = row[i0 + 3];
    float ss = v0 * v0 + v1 * v1 + v2 * v2 + v3 * v3;
#pragma unroll
    for (int m = 16; m > 0; m >>= 1) ss += __shfl_xor(ss, m, 32);
    if (lane == 0) red[wave] = ss;
    __syncthreads();
    float tot = 0.0f;
#pragma unroll
    for (int j = 0; j < 8; ++j) tot += red[j];
    float scale = rsqrtf(tot * (1.0f / (float)CC) + 1e-6f);
    row[i0]     = v0 * scale * w[i0];
    row[i0 + 1] = v1 * scale * w[i0 + 1];
    row[i0 + 2] = v2 * scale * w[i0 + 2];
    row[i0 + 3] = v3 * scale * w[i0 + 3];
    __syncthreads();
  }
}

// ---------------- small-world attention: one wave per (token, head) --------

__global__ __launch_bounds__(256) void attn_kernel(const float* __restrict__ qkv,
                                                   const float* __restrict__ eb,
                                                   bf16* __restrict__ A) {
  const int SP[12] = {1, -1, 2, -2, 4, -4, 8, -8, 16, -16, 32, -32};
  const int TP[4]  = {1, -1, 2, -2};
  const int lane = threadIdx.x & 31;
  const int wid = (blockIdx.x << 3) + (threadIdx.x >> 5);  // 8 waves / block
  const int l = wid >> 4;        // token
  const int h = wid & 15;        // head
  const int t = l >> 10;         // frame
  const int s = l & (SS - 1);    // spatial idx
  const int d0 = lane << 1;      // 2 elems of D=64 per lane

  const size_t hoff = (size_t)h * HD + d0;
  const float qx = qkv[(size_t)l * NQKV + hoff];
  const float qy = qkv[(size_t)l * NQKV + hoff + 1];

  int nb[NSH];
#pragma unroll
  for (int n = 0; n < 12; ++n) nb[n] = (t << 10) + ((s + SP[n]) & (SS - 1));
#pragma unroll
  for (int n = 0; n < 4; ++n) nb[12 + n] = (((t + TP[n]) & (TT - 1)) << 10) + s;

  float sc[NSH];
#pragma unroll
  for (int n = 0; n < NSH; ++n) {
    size_t off = (size_t)nb[n] * NQKV + CC + hoff;   // K block
    float p = qx * qkv[off] + qy * qkv[off + 1];
#pragma unroll
    for (int m = 16; m > 0; m >>= 1) p += __shfl_xor(p, m, 32);
    sc[n] = p * 0.125f + eb[h * NSH + n];            // scale = D^-0.5
  }

  float mx = sc[0];
#pragma unroll
  for (int n = 1; n < NSH; ++n) mx = fmaxf(mx, sc[n]);
  float sum = 0.0f;
#pragma unroll
  for (int n = 0; n < NSH; ++n) { sc[n] = __expf(sc[n] - mx); sum += sc[n]; }
  const float inv = 1.0f / sum;

  float ax = 0.0f, ay = 0.0f;
#pragma unroll
  for (int n = 0; n < NSH; ++n) {
    size_t off = (size_t)nb[n] * NQKV + 2 * CC + hoff;  // V block
    float w = sc[n] * inv;
    ax += w * qkv[off];
    ay += w * qkv[off + 1];
  }
  bf16* dst = A + (size_t)l * CC + hoff;
  dst[0] = (bf16)ax;
  dst[1] = (bf16)ay;
}

// ---------------- host-side orchestration ----------------------------------

extern "C" void kernel_launch(void* const* d_in, const int* in_sizes, int n_in,
                              void* d_out, int out_size, void* d_ws, size_t ws_size,
                              hipStream_t stream) {
  (void)in_sizes; (void)n_in; (void)out_size; (void)ws_size;
  const float* x    = (const float*)d_in[0];
  const float* Wq   = (const float*)d_in[1];
  const float* bq   = (const float*)d_in[2];
  const float* Wk   = (const float*)d_in[3];
  const float* bk   = (const float*)d_in[4];
  const float* Wv   = (const float*)d_in[5];
  const float* bv   = (const float*)d_in[6];
  const float* qn_w = (const float*)d_in[7];
  const float* kn_w = (const float*)d_in[8];
  const float* eb   = (const float*)d_in[9];
  const float* Wo   = (const float*)d_in[10];
  const float* bo   = (const float*)d_in[11];
  float* out = (float*)d_out;

  char* ws = (char*)d_ws;
  size_t off = 0;
  auto alloc = [&](size_t bytes) -> void* {
    void* p = ws + off;
    off = (off + bytes + 255) & ~(size_t)255;
    return p;
  };
  bf16*  xb    = (bf16*)alloc((size_t)LL * CC * 2);        // 16 MB
  bf16*  wqkvT = (bf16*)alloc((size_t)CC * NQKV * 2);      //  6 MB, [3072][1024]
  bf16*  woT   = (bf16*)alloc((size_t)CC * CC * 2);        //  2 MB, [1024][1024]
  float* bqkv  = (float*)alloc((size_t)NQKV * 4);          // 12 KB
  float* qkv   = (float*)alloc((size_t)LL * NQKV * 4);     // 96 MB
  bf16*  abuf  = (bf16*)alloc((size_t)LL * CC * 2);        // 16 MB

  // 1. convert / pack (weights transposed so GEMM B staging is b128-clean)
  conv_f32_to_bf16<<<(LL * CC) / 1024, 256, 0, stream>>>(x, xb, LL * CC);
  pack_wqkvT<<<(3 * CC * CC) / 256, 256, 0, stream>>>(Wq, Wk, Wv, wqkvT);
  pack_bqkv<<<NQKV / 256, 256, 0, stream>>>(bq, bk, bv, bqkv);
  pack_woT<<<(CC * CC) / 256, 256, 0, stream>>>(Wo, woT);

  // 2. QKV projection: [8192,1024] @ [1024,3072] + bias
  gemm_bf16_wmma<<<dim3(NQKV / 64, LL / 256), 256, 0, stream>>>(
      xb, wqkvT, bqkv, qkv, LL, NQKV, CC);

  // 3. RMS-normalize Q and K in place
  rms_kernel<<<LL, 256, 0, stream>>>(qkv, qn_w, kn_w);

  // 4. small-world attention -> bf16 activations
  attn_kernel<<<(LL * NH) / 8, 256, 0, stream>>>(qkv, eb, abuf);

  // 5. output projection: [8192,1024] @ [1024,1024] + bo -> d_out (f32)
  gemm_bf16_wmma<<<dim3(CC / 64, LL / 256), 256, 0, stream>>>(
      abuf, woT, bo, out, LL, CC, CC);
}
